// MultiheadAttnAndTransition_47519518163043
// MI455X (gfx1250) — matrix operand order repaired and checked by reading, not compiled, not run on hardware
//
#include <hip/hip_runtime.h>
#include <cstdint>
#include <cstddef>

// ---------------------------------------------------------------------------
// MI455X (gfx1250) implementation of the windowed-attention + transition block.
// - All GEMMs on v_wmma_f32_16x16x32_bf16 (wave32 WMMA, f32 accum).
// - Weights pre-converted once per launch to bf16 [N][K]; GEMM tiles are then
//   staged with global_load_async_to_lds_b128 (ASYNCcnt) and double-buffered.
// - Pair bias computed only at gathered window key columns (64 MB vs 256 MB).
// Whole block is HBM-bound (~150 MB -> ~7 us at 23.3 TB/s).
// ---------------------------------------------------------------------------

typedef __attribute__((ext_vector_type(16))) __bf16 v16bf;
typedef __attribute__((ext_vector_type(8)))  float  v8f;

#define B_     2
#define N_     512
#define H_     12
#define DH_    64
#define DIM_   768
#define DPAIR_ 128
#define DCOND_ 512
#define INNER_ 3072
#define NW_    16     // N_ / n_queries
#define NQ_    32     // n_queries (fixed by setup_inputs)
#define NK_    128    // n_keys    (fixed by setup_inputs)
#define TOK_   (B_ * N_)   // 1024 tokens

#define DEV __device__ __forceinline__

DEV float sigmoidf_(float x) { return 1.0f / (1.0f + __expf(-x)); }

DEV float wave_sum32(float v) {
#pragma unroll
  for (int m = 16; m >= 1; m >>= 1) v += __shfl_xor(v, m, 32);
  return v;
}

DEV float block_sum256(float v, float* red) {
  v = wave_sum32(v);
  if ((threadIdx.x & 31) == 0) red[threadIdx.x >> 5] = v;
  __syncthreads();
  float tot = 0.f;
#pragma unroll
  for (int i = 0; i < 8; ++i) tot += red[i];
  __syncthreads();
  return tot;
}

// Async 16B global->LDS copy (gfx1250, tracked by ASYNCcnt, bypasses VGPRs).
DEV void async_copy16(unsigned lds_off, const void* gsrc) {
  asm volatile("global_load_async_to_lds_b128 %0, %1, off"
               :: "v"(lds_off), "v"((unsigned long long)(size_t)gsrc)
               : "memory");
}

// ---- WMMA fragment loads (wave32); contiguous k = 16*(lane/16) + elem in A and B --------------------
// A (16x32 bf16), LDS tile row-major [row][K], stride in halves.
// lanes 0-15: elems 0..15 = K 0..15 (row = lane)
// lanes 16-31: elems 0..15 = K 16..31 (row = lane-16)
DEV v16bf wmma_load_A(const __bf16* lds, int row0, int stride, int ktile) {
  int lane = threadIdx.x & 31;
  const __bf16* p = lds + (size_t)(row0 + (lane & 15)) * stride + ktile;
  int kb = (lane >> 4) * 16;
  v16bf a;
#pragma unroll
  for (int i = 0; i < 8; ++i) a[i] = p[kb + i];
#pragma unroll
  for (int i = 0; i < 8; ++i) a[8 + i] = p[kb + 8 + i];
  return a;
}

// B (32x16 bf16) from LDS tile stored transposed [N][K] (row-major, stride in
// halves). lane holds column N = (lane&15); elems j = K (lane>>4)*16 + j.
DEV v16bf wmma_load_Bt(const __bf16* lds, int col0, int stride, int ktile) {
  int lane = threadIdx.x & 31;
  const __bf16* p = lds + (size_t)(col0 + (lane & 15)) * stride + ktile + (lane >> 4) * 16;
  v16bf b;
#pragma unroll
  for (int i = 0; i < 16; ++i) b[i] = p[i];
  return b;
}

// ---------------------------------------------------------------------------
// Weight pre-pass: W[K][N] f32 -> Wt[N][K] bf16 (LDS-tiled transpose).
// ---------------------------------------------------------------------------
__global__ __launch_bounds__(256) void xpose_bf16_kernel(const float* __restrict__ W,
                                                         __bf16* __restrict__ Wt,
                                                         int K, int N) {
  __shared__ float tile[32][33];
  int k0 = blockIdx.x * 32, n0 = blockIdx.y * 32;
  int tx = threadIdx.x & 31, ty = threadIdx.x >> 5;   // 32 x 8
#pragma unroll
  for (int r = ty; r < 32; r += 8)
    tile[r][tx] = W[(size_t)(k0 + r) * N + n0 + tx];
  __syncthreads();
#pragma unroll
  for (int r = ty; r < 32; r += 8)
    Wt[(size_t)(n0 + r) * K + k0 + tx] = (__bf16)tile[tx][r];
}

// ---------------------------------------------------------------------------
// GEMM: C[M x N] = A_bf16[M x K] @ Wt_bf16[N x K]^T (+ bias[N])
// 64x64 block tile, BK=32, 8 waves -> two 16x16 D tiles per wave.
// Tiles staged with async global->LDS copies, double-buffered:
//   issue stage k+1 ; s_wait_asynccnt<=2 (stage k landed, in-order) ; barrier ;
//   WMMA on stage k ; barrier (protects buffer from stage k+2 overwrite).
// M,K,N multiples of 64/32/64 in all uses.
// ---------------------------------------------------------------------------
__global__ __launch_bounds__(256) void gemm_bf16(const __bf16* __restrict__ A,
                                                 const __bf16* __restrict__ Wt,
                                                 const float* __restrict__ bias,
                                                 float* __restrict__ C,
                                                 int M, int N, int K) {
  __shared__ __align__(16) __bf16 As[2][64][32];
  __shared__ __align__(16) __bf16 Bts[2][64][32];   // [n][k]
  const int bm = blockIdx.x * 64, bn = blockIdx.y * 64;
  const int t = threadIdx.x, wid = t >> 5, lane = t & 31;
  const int mt = wid & 3, ng = wid >> 2;
  const int row = t >> 2, c8 = (t & 3) * 8;   // 16B per thread per tile
  const __bf16* gA = A + (size_t)(bm + row) * K + c8;
  const __bf16* gB = Wt + (size_t)(bn + row) * K + c8;
  unsigned ldsA0 = (unsigned)(size_t)&As[0][row][c8];
  unsigned ldsA1 = (unsigned)(size_t)&As[1][row][c8];
  unsigned ldsB0 = (unsigned)(size_t)&Bts[0][row][c8];
  unsigned ldsB1 = (unsigned)(size_t)&Bts[1][row][c8];

  // prologue: stage 0
  async_copy16(ldsA0, gA);
  async_copy16(ldsB0, gB);

  v8f acc0 = {}; v8f acc1 = {};
  const int nk = K / 32;
  for (int kb = 0; kb < nk; ++kb) {
    int cur = kb & 1;
    if (kb + 1 < nk) {
      async_copy16(cur ? ldsA0 : ldsA1, gA + (kb + 1) * 32);
      async_copy16(cur ? ldsB0 : ldsB1, gB + (kb + 1) * 32);
      asm volatile("s_wait_asynccnt 0x2" ::: "memory");
    } else {
      asm volatile("s_wait_asynccnt 0x0" ::: "memory");
    }
    __syncthreads();
    v16bf a  = wmma_load_A(&As[cur][0][0], mt * 16, 32, 0);
    v16bf b0 = wmma_load_Bt(&Bts[cur][0][0], (ng * 2) * 16, 32, 0);
    v16bf b1 = wmma_load_Bt(&Bts[cur][0][0], (ng * 2 + 1) * 16, 32, 0);
    acc0 = __builtin_amdgcn_wmma_f32_16x16x32_bf16(false, a, false, b0, (short)0, acc0, false, false);
    acc1 = __builtin_amdgcn_wmma_f32_16x16x32_bf16(false, a, false, b1, (short)0, acc1, false, false);
    __syncthreads();
  }
  int col = lane & 15, rb = (lane >> 4) * 8;
#pragma unroll
  for (int s = 0; s < 2; ++s) {
    v8f acc = s ? acc1 : acc0;
    int gc = bn + (ng * 2 + s) * 16 + col;
    float bv = bias ? bias[gc] : 0.f;
#pragma unroll
    for (int i = 0; i < 8; ++i) {
      int gr = bm + mt * 16 + rb + i;
      C[(size_t)gr * N + gc] = acc[i] + bv;
    }
  }
}

// ---------------------------------------------------------------------------
// cn = LN(cond) -> bf16, one block per token row (512 cols, 2 per thread)
// ---------------------------------------------------------------------------
__global__ __launch_bounds__(256) void cond_ln_kernel(const float* __restrict__ cond,
                                                      __bf16* __restrict__ cnb) {
  __shared__ float red[8];
  size_t base = (size_t)blockIdx.x * DCOND_;
  float v0 = cond[base + threadIdx.x], v1 = cond[base + threadIdx.x + 256];
  float mean = block_sum256(v0 + v1, red) * (1.f / 512.f);
  float d0 = v0 - mean, d1 = v1 - mean;
  float var = block_sum256(d0 * d0 + d1 * d1, red) * (1.f / 512.f);
  float r = rsqrtf(var + 1e-5f);
  cnb[base + threadIdx.x]       = (__bf16)(d0 * r);
  cnb[base + threadIdx.x + 256] = (__bf16)(d1 * r);
}

// ---------------------------------------------------------------------------
// xa = (LN(x*mask) * sigmoid(g1) + b1) * mask  -> bf16 (768 cols, 3/thread)
// ---------------------------------------------------------------------------
__global__ __launch_bounds__(256) void xa_kernel(const float* __restrict__ x,
                                                 const float* __restrict__ mask,
                                                 const float* __restrict__ g1,
                                                 const float* __restrict__ b1,
                                                 __bf16* __restrict__ xab) {
  __shared__ float red[8];
  float m = mask[blockIdx.x];
  size_t base = (size_t)blockIdx.x * DIM_;
  float v[3];
#pragma unroll
  for (int j = 0; j < 3; ++j) v[j] = x[base + threadIdx.x + j * 256] * m;
  float mean = block_sum256(v[0] + v[1] + v[2], red) * (1.f / 768.f);
  float d[3];
#pragma unroll
  for (int j = 0; j < 3; ++j) d[j] = v[j] - mean;
  float var = block_sum256(d[0]*d[0] + d[1]*d[1] + d[2]*d[2], red) * (1.f / 768.f);
  float r = rsqrtf(var + 1e-5f);
#pragma unroll
  for (int j = 0; j < 3; ++j) {
    size_t c = base + threadIdx.x + j * 256;
    xab[c] = (__bf16)(((d[j] * r) * sigmoidf_(g1[c]) + b1[c]) * m);
  }
}

// ---------------------------------------------------------------------------
// per-head LN + RoPE on q,k; v -> bf16.  One wave per (b,n,h), lane = pair.
// ---------------------------------------------------------------------------
__global__ __launch_bounds__(32) void qk_rope_kernel(const float* __restrict__ qf,
                                                     const float* __restrict__ kf,
                                                     const float* __restrict__ vf,
                                                     __bf16* __restrict__ qb,
                                                     __bf16* __restrict__ kb,
                                                     __bf16* __restrict__ vb) {
  int bid = blockIdx.x;
  int h = bid % H_;
  int row = bid / H_;           // b*N + n
  int n = row & (N_ - 1);
  size_t base = (size_t)row * DIM_ + h * DH_;
  int lane = threadIdx.x;
  float inv = powf(10000.f, -(2.f * lane) / (float)DH_);
  float ang = (float)n * inv;
  float cs = __cosf(ang), sn = __sinf(ang);
#pragma unroll
  for (int which = 0; which < 2; ++which) {
    const float* src = which ? kf : qf;
    __bf16* dst = which ? kb : qb;
    float e0 = src[base + 2 * lane], e1 = src[base + 2 * lane + 1];
    float mean = wave_sum32(e0 + e1) * (1.f / 64.f);
    float d0 = e0 - mean, d1 = e1 - mean;
    float var = wave_sum32(d0 * d0 + d1 * d1) * (1.f / 64.f);
    float r = rsqrtf(var + 1e-5f);
    d0 *= r; d1 *= r;
    dst[base + 2 * lane]     = (__bf16)(d0 * cs - d1 * sn);
    dst[base + 2 * lane + 1] = (__bf16)(d0 * sn + d1 * cs);
  }
  vb[base + 2 * lane]     = (__bf16)vf[base + 2 * lane];
  vb[base + 2 * lane + 1] = (__bf16)vf[base + 2 * lane + 1];
}

// ---------------------------------------------------------------------------
// Windowed pair bias: biasw[b,h,w,q,kk] = LN(pair_rep[b, i, jc, :]) . wbp[:,h]
// Only the 128 gathered key columns per query row (64 MB instead of 256 MB).
// One block per (b, gi); one wave per key row; lane holds 4 channels.
// ---------------------------------------------------------------------------
__global__ __launch_bounds__(256) void bias_pair_kernel(const float* __restrict__ pair_rep,
                                                        const float* __restrict__ wbp,
                                                        float* __restrict__ biasw) {
  __shared__ float wsm[DPAIR_ * H_];
  int bi = blockIdx.x;
  int b = bi >> 9, gi = bi & (N_ - 1);
  int w = gi >> 5, qq = gi & (NQ_ - 1);
  for (int i = threadIdx.x; i < DPAIR_ * H_; i += 256) wsm[i] = wbp[i];
  __syncthreads();
  int wid = threadIdx.x >> 5, lane = threadIdx.x & 31;
  for (int kk = wid; kk < NK_; kk += 8) {
    int idx = w * NQ_ - NK_ / 2 + NQ_ / 2 + kk;   // center - 64 + kk
    int jc = min(max(idx, 0), N_ - 1);
    const float* src = pair_rep + ((size_t)(b * N_ + gi) * N_ + jc) * DPAIR_;
    float4 xv = *(const float4*)(src + lane * 4);
    float mean = wave_sum32(xv.x + xv.y + xv.z + xv.w) * (1.f / 128.f);
    float d0 = xv.x - mean, d1 = xv.y - mean, d2 = xv.z - mean, d3 = xv.w - mean;
    float var = wave_sum32(d0*d0 + d1*d1 + d2*d2 + d3*d3) * (1.f / 128.f);
    float r = rsqrtf(var + 1e-5f);
    d0 *= r; d1 *= r; d2 *= r; d3 *= r;
    const float* w0 = &wsm[(lane * 4) * H_];
    float acc[H_];
#pragma unroll
    for (int h = 0; h < H_; ++h)
      acc[h] = d0 * w0[h] + d1 * w0[H_ + h] + d2 * w0[2 * H_ + h] + d3 * w0[3 * H_ + h];
#pragma unroll
    for (int m = 16; m >= 1; m >>= 1)
#pragma unroll
      for (int h = 0; h < H_; ++h) acc[h] += __shfl_xor(acc[h], m, 32);
    if (lane < H_) {
      float outv = 0.f;
#pragma unroll
      for (int h = 0; h < H_; ++h) if (lane == h) outv = acc[h];
      biasw[((((size_t)b * H_ + lane) * NW_ + w) * NQ_ + qq) * NK_ + kk] = outv;
    }
  }
}

// ---------------------------------------------------------------------------
// Windowed attention, one block per (b,h,window). WMMA scores -> softmax ->
// WMMA out; epilogue fuses the sigmoid gate and writes bf16 for the wo GEMM.
// ---------------------------------------------------------------------------
__global__ __launch_bounds__(256) void attn_kernel(const __bf16* __restrict__ qb,
                                                   const __bf16* __restrict__ kb,
                                                   const __bf16* __restrict__ vb,
                                                   const float* __restrict__ biasw,
                                                   const float* __restrict__ pair_mask,
                                                   const float* __restrict__ gproj,
                                                   __bf16* __restrict__ ogb) {
  __shared__ __align__(16) __bf16 qs[NQ_][DH_];
  __shared__ __align__(16) __bf16 ks[NK_][DH_];
  __shared__ __align__(16) __bf16 vts[DH_][132];   // [d][key], padded stride
  __shared__ float Ss[NQ_][129];
  __shared__ __align__(16) __bf16 ps[NQ_][132];
  __shared__ int jcs[NK_];
  __shared__ int vvd[NK_];

  int blk = blockIdx.x;
  int b = blk / (H_ * NW_);
  int rem = blk % (H_ * NW_);
  int h = rem / NW_;
  int w = rem % NW_;
  int t = threadIdx.x, wid = t >> 5, lane = t & 31;

  { // q: 32 rows x 64 halves
    int row = t >> 3, c8 = (t & 7) * 8;
    const __bf16* src = qb + (size_t)(b * N_ + w * NQ_ + row) * DIM_ + h * DH_ + c8;
    *(uint4*)&qs[row][c8] = *(const uint4*)src;
  }
  { // k, v: 128 gathered rows x 64 halves (2 threads per row)
    int row = t >> 1, half = (t & 1) * 32;
    int idx = w * NQ_ - NK_ / 2 + NQ_ / 2 + row;
    int jc = min(max(idx, 0), N_ - 1);
    if (half == 0) { jcs[row] = jc; vvd[row] = (idx >= 0 && idx < N_) ? 1 : 0; }
    const __bf16* ksrc = kb + (size_t)(b * N_ + jc) * DIM_ + h * DH_ + half;
    const __bf16* vsrc = vb + (size_t)(b * N_ + jc) * DIM_ + h * DH_ + half;
#pragma unroll
    for (int c = 0; c < 4; ++c)
      *(uint4*)&ks[row][half + c * 8] = *(const uint4*)(ksrc + c * 8);
#pragma unroll
    for (int d = 0; d < 32; ++d) vts[half + d][row] = vsrc[d];
  }
  __syncthreads();

  { // scores S = q @ k^T / 8 + bias; 2x8 tiles, 2 per wave
    int mt = wid >> 2;
#pragma unroll
    for (int s = 0; s < 2; ++s) {
      int nt = (wid & 3) * 2 + s;
      v8f acc = {};
#pragma unroll
      for (int kt = 0; kt < DH_; kt += 32) {
        v16bf a  = wmma_load_A(&qs[0][0], mt * 16, DH_, kt);
        v16bf bb = wmma_load_Bt(&ks[0][0], nt * 16, DH_, kt);
        acc = __builtin_amdgcn_wmma_f32_16x16x32_bf16(false, a, false, bb, (short)0, acc, false, false);
      }
      int col = nt * 16 + (lane & 15);
      int rb = mt * 16 + (lane >> 4) * 8;
      int jc = jcs[col], vd = vvd[col];
#pragma unroll
      for (int i = 0; i < 8; ++i) {
        int row = rb + i, gi = w * NQ_ + row;
        float pm = pair_mask[(size_t)b * N_ * N_ + (size_t)gi * N_ + jc];
        float sc = acc[i] * 0.125f +
                   biasw[((((size_t)b * H_ + h) * NW_ + w) * NQ_ + row) * NK_ + col];
        Ss[row][col] = (vd && pm > 0.f) ? sc : -1e9f;
      }
    }
  }
  __syncthreads();

  if (t < NQ_) { // row softmax -> bf16 probabilities
    float mx = -1e30f;
    for (int c = 0; c < NK_; ++c) mx = fmaxf(mx, Ss[t][c]);
    float sum = 0.f;
    for (int c = 0; c < NK_; ++c) sum += __expf(Ss[t][c] - mx);
    float rs = 1.f / sum;
    for (int c = 0; c < NK_; ++c) ps[t][c] = (__bf16)(__expf(Ss[t][c] - mx) * rs);
  }
  __syncthreads();

  { // out = P @ V; 2x4 tiles, 1 per wave; fuse gate, write bf16
    int mt = wid & 1, nt = wid >> 1;
    v8f acc = {};
#pragma unroll
    for (int kt = 0; kt < NK_; kt += 32) {
      v16bf a  = wmma_load_A(&ps[0][0], mt * 16, 132, kt);
      v16bf bb = wmma_load_Bt(&vts[0][0], nt * 16, 132, kt);
      acc = __builtin_amdgcn_wmma_f32_16x16x32_bf16(false, a, false, bb, (short)0, acc, false, false);
    }
    int dcol = nt * 16 + (lane & 15);
    int rb = mt * 16 + (lane >> 4) * 8;
#pragma unroll
    for (int i = 0; i < 8; ++i) {
      int row = rb + i, gi = w * NQ_ + row;
      size_t gp = (size_t)(b * N_ + gi) * DIM_ + h * DH_ + dcol;
      ogb[gp] = (__bf16)(acc[i] * sigmoidf_(gproj[gp]));
    }
  }
}

// ---------------------------------------------------------------------------
// post-attn: scale1 + residual, then adaLN2 -> xt (bf16); stores x2 (f32).
// ---------------------------------------------------------------------------
__global__ __launch_bounds__(256) void post_attn_kernel(const float* __restrict__ x,
                                                        const float* __restrict__ mask,
                                                        const float* __restrict__ aproj,
                                                        const float* __restrict__ s1,
                                                        const float* __restrict__ g2,
                                                        const float* __restrict__ b2,
                                                        float* __restrict__ x2,
                                                        __bf16* __restrict__ xtb) {
  __shared__ float red[8];
  float m = mask[blockIdx.x];
  size_t base = (size_t)blockIdx.x * DIM_;
  float v[3];
#pragma unroll
  for (int j = 0; j < 3; ++j) {
    size_t c = base + threadIdx.x + j * 256;
    float ao = aproj[c] * sigmoidf_(s1[c]) * m;  // _scale_out (incl. mask)
    ao *= m;                                     // extra mask in reference
    float xv = (ao + x[c] * m) * m;
    x2[c] = xv;
    v[j] = xv;
  }
  float mean = block_sum256(v[0] + v[1] + v[2], red) * (1.f / 768.f);
  float d[3];
#pragma unroll
  for (int j = 0; j < 3; ++j) d[j] = v[j] - mean;
  float var = block_sum256(d[0]*d[0] + d[1]*d[1] + d[2]*d[2], red) * (1.f / 768.f);
  float r = rsqrtf(var + 1e-5f);
#pragma unroll
  for (int j = 0; j < 3; ++j) {
    size_t c = base + threadIdx.x + j * 256;
    xtb[c] = (__bf16)(((d[j] * r) * sigmoidf_(g2[c]) + b2[c]) * m);
  }
}

__global__ __launch_bounds__(256) void swiglu_kernel(const float* __restrict__ t1,
                                                     __bf16* __restrict__ hb) {
  int total = TOK_ * INNER_;
  for (int i = blockIdx.x * blockDim.x + threadIdx.x; i < total; i += gridDim.x * blockDim.x) {
    int row = i / INNER_, c = i % INNER_;
    float a = t1[(size_t)row * (2 * INNER_) + c];
    float g = t1[(size_t)row * (2 * INNER_) + INNER_ + c];
    hb[i] = (__bf16)(a * g * sigmoidf_(g));
  }
}

__global__ __launch_bounds__(256) void final_kernel(const float* __restrict__ t2,
                                                    const float* __restrict__ s2,
                                                    const float* __restrict__ x2,
                                                    const float* __restrict__ mask,
                                                    float* __restrict__ outp) {
  int total = TOK_ * DIM_;
  for (int i = blockIdx.x * blockDim.x + threadIdx.x; i < total; i += gridDim.x * blockDim.x) {
    int row = i / DIM_;
    float m = mask[row];
    float tr = t2[i] * m;                 // (a*silu(g)) @ tr_wout * mask
    tr = tr * sigmoidf_(s2[i]) * m;       // _scale_out
    tr *= m;                              // extra mask
    outp[i] = ((tr + x2[i]) * m) * m;     // residual, final mask
  }
}

// ---------------------------------------------------------------------------
extern "C" void kernel_launch(void* const* d_in, const int* in_sizes, int n_in,
                              void* d_out, int out_size, void* d_ws, size_t ws_size,
                              hipStream_t stream) {
  (void)in_sizes; (void)n_in; (void)out_size; (void)ws_size;
  const float* x      = (const float*)d_in[0];
  const float* prep   = (const float*)d_in[1];
  const float* cond   = (const float*)d_in[2];
  const float* mask   = (const float*)d_in[3];
  const float* pmask  = (const float*)d_in[4];
  const float* a1_gw  = (const float*)d_in[5];
  const float* a1_gb  = (const float*)d_in[6];
  const float* a1_bw  = (const float*)d_in[7];
  const float* wq     = (const float*)d_in[8];
  const float* bq     = (const float*)d_in[9];
  const float* wk     = (const float*)d_in[10];
  const float* bk     = (const float*)d_in[11];
  const float* wvw    = (const float*)d_in[12];
  const float* bvv    = (const float*)d_in[13];
  const float* wg     = (const float*)d_in[14];
  const float* bg     = (const float*)d_in[15];
  const float* wbp    = (const float*)d_in[16];
  const float* wo     = (const float*)d_in[17];
  const float* bo     = (const float*)d_in[18];
  const float* s1w    = (const float*)d_in[19];
  const float* s1b    = (const float*)d_in[20];
  const float* a2_gw  = (const float*)d_in[21];
  const float* a2_gb  = (const float*)d_in[22];
  const float* a2_bw  = (const float*)d_in[23];
  const float* trwin  = (const float*)d_in[24];
  const float* trwout = (const float*)d_in[25];
  const float* s2w    = (const float*)d_in[26];
  const float* s2b    = (const float*)d_in[27];
  // d_in[28]=n_queries(32), d_in[29]=n_keys(128): compile-time constants here.
  float* outp = (float*)d_out;

  char* p = (char*)d_ws;
  auto alloc = [&](size_t bytes) -> void* {
    void* r = (void*)p;
    p += (bytes + 255) & ~(size_t)255;
    return r;
  };
  // activations / intermediates
  __bf16* cnb   = (__bf16*)alloc((size_t)TOK_ * DCOND_ * 2);
  float* g1     = (float*)alloc((size_t)TOK_ * DIM_ * 4);
  float* b1     = (float*)alloc((size_t)TOK_ * DIM_ * 4);
  float* s1     = (float*)alloc((size_t)TOK_ * DIM_ * 4);
  float* g2     = (float*)alloc((size_t)TOK_ * DIM_ * 4);
  float* b2     = (float*)alloc((size_t)TOK_ * DIM_ * 4);
  float* s2     = (float*)alloc((size_t)TOK_ * DIM_ * 4);
  __bf16* xab   = (__bf16*)alloc((size_t)TOK_ * DIM_ * 2);
  float* qf     = (float*)alloc((size_t)TOK_ * DIM_ * 4);
  float* kf     = (float*)alloc((size_t)TOK_ * DIM_ * 4);
  float* vf     = (float*)alloc((size_t)TOK_ * DIM_ * 4);
  float* gf     = (float*)alloc((size_t)TOK_ * DIM_ * 4);
  __bf16* qbb   = (__bf16*)alloc((size_t)TOK_ * DIM_ * 2);
  __bf16* kbb   = (__bf16*)alloc((size_t)TOK_ * DIM_ * 2);
  __bf16* vbb   = (__bf16*)alloc((size_t)TOK_ * DIM_ * 2);
  float* biasw  = (float*)alloc((size_t)B_ * H_ * NW_ * NQ_ * NK_ * 4);
  __bf16* ogb   = (__bf16*)alloc((size_t)TOK_ * DIM_ * 2);
  float* aproj  = (float*)alloc((size_t)TOK_ * DIM_ * 4);
  float* x2     = (float*)alloc((size_t)TOK_ * DIM_ * 4);
  __bf16* xtb   = (__bf16*)alloc((size_t)TOK_ * DIM_ * 2);
  float* t1     = (float*)alloc((size_t)TOK_ * 2 * INNER_ * 4);
  __bf16* hb    = (__bf16*)alloc((size_t)TOK_ * INNER_ * 2);
  float* t2     = (float*)alloc((size_t)TOK_ * DIM_ * 4);
  // pre-transposed bf16 weights [N][K]
  __bf16* t_a1gw = (__bf16*)alloc((size_t)DIM_ * DCOND_ * 2);
  __bf16* t_a1bw = (__bf16*)alloc((size_t)DIM_ * DCOND_ * 2);
  __bf16* t_s1w  = (__bf16*)alloc((size_t)DIM_ * DCOND_ * 2);
  __bf16* t_a2gw = (__bf16*)alloc((size_t)DIM_ * DCOND_ * 2);
  __bf16* t_a2bw = (__bf16*)alloc((size_t)DIM_ * DCOND_ * 2);
  __bf16* t_s2w  = (__bf16*)alloc((size_t)DIM_ * DCOND_ * 2);
  __bf16* t_wq   = (__bf16*)alloc((size_t)DIM_ * DIM_ * 2);
  __bf16* t_wk   = (__bf16*)alloc((size_t)DIM_ * DIM_ * 2);
  __bf16* t_wv   = (__bf16*)alloc((size_t)DIM_ * DIM_ * 2);
  __bf16* t_wg   = (__bf16*)alloc((size_t)DIM_ * DIM_ * 2);
  __bf16* t_wo   = (__bf16*)alloc((size_t)DIM_ * DIM_ * 2);
  __bf16* t_win  = (__bf16*)alloc((size_t)(2 * INNER_) * DIM_ * 2);
  __bf16* t_wout = (__bf16*)alloc((size_t)DIM_ * INNER_ * 2);

  auto xpose = [&](const float* W, __bf16* Wt, int K, int N) {
    dim3 g(K / 32, N / 32);
    xpose_bf16_kernel<<<g, 256, 0, stream>>>(W, Wt, K, N);
  };

  // 0) weight pre-pass (bf16, transposed to [N][K])
  xpose(a1_gw, t_a1gw, DCOND_, DIM_);
  xpose(a1_bw, t_a1bw, DCOND_, DIM_);
  xpose(s1w,   t_s1w,  DCOND_, DIM_);
  xpose(a2_gw, t_a2gw, DCOND_, DIM_);
  xpose(a2_bw, t_a2bw, DCOND_, DIM_);
  xpose(s2w,   t_s2w,  DCOND_, DIM_);
  xpose(wq,    t_wq,   DIM_, DIM_);
  xpose(wk,    t_wk,   DIM_, DIM_);
  xpose(wvw,   t_wv,   DIM_, DIM_);
  xpose(wg,    t_wg,   DIM_, DIM_);
  xpose(wo,    t_wo,   DIM_, DIM_);
  xpose(trwin, t_win,  DIM_, 2 * INNER_);
  xpose(trwout, t_wout, INNER_, DIM_);

  // 1) cn = LN(cond)
  cond_ln_kernel<<<TOK_, 256, 0, stream>>>(cond, cnb);

  // 2) six conditioning projections (K = 512)
  dim3 gC(TOK_ / 64, DIM_ / 64);
  gemm_bf16<<<gC, 256, 0, stream>>>(cnb, t_a1gw, a1_gb, g1, TOK_, DIM_, DCOND_);
  gemm_bf16<<<gC, 256, 0, stream>>>(cnb, t_a1bw, nullptr, b1, TOK_, DIM_, DCOND_);
  gemm_bf16<<<gC, 256, 0, stream>>>(cnb, t_s1w,  s1b,     s1, TOK_, DIM_, DCOND_);
  gemm_bf16<<<gC, 256, 0, stream>>>(cnb, t_a2gw, a2_gb,  g2, TOK_, DIM_, DCOND_);
  gemm_bf16<<<gC, 256, 0, stream>>>(cnb, t_a2bw, nullptr, b2, TOK_, DIM_, DCOND_);
  gemm_bf16<<<gC, 256, 0, stream>>>(cnb, t_s2w,  s2b,     s2, TOK_, DIM_, DCOND_);

  // 3) adaLN1 -> xa (bf16)
  xa_kernel<<<TOK_, 256, 0, stream>>>(x, mask, g1, b1, xab);

  // 4) QKVG projections (K = 768)
  gemm_bf16<<<gC, 256, 0, stream>>>(xab, t_wq, bq,  qf, TOK_, DIM_, DIM_);
  gemm_bf16<<<gC, 256, 0, stream>>>(xab, t_wk, bk,  kf, TOK_, DIM_, DIM_);
  gemm_bf16<<<gC, 256, 0, stream>>>(xab, t_wv, bvv, vf, TOK_, DIM_, DIM_);
  gemm_bf16<<<gC, 256, 0, stream>>>(xab, t_wg, bg,  gf, TOK_, DIM_, DIM_);

  // 5) per-head LN + RoPE (q,k) and bf16 convert (v)
  qk_rope_kernel<<<TOK_ * H_, 32, 0, stream>>>(qf, kf, vf, qbb, kbb, vbb);

  // 6) windowed pair bias (gathered columns only)
  bias_pair_kernel<<<TOK_, 256, 0, stream>>>(prep, wbp, biasw);

  // 7) windowed attention with fused sigmoid gate
  attn_kernel<<<B_ * H_ * NW_, 256, 0, stream>>>(qbb, kbb, vbb, biasw, pmask, gf, ogb);

  // 8) output projection
  gemm_bf16<<<gC, 256, 0, stream>>>(ogb, t_wo, bo, aproj, TOK_, DIM_, DIM_);

  // 9) scale1 + residual + adaLN2
  post_attn_kernel<<<TOK_, 256, 0, stream>>>(x, mask, aproj, s1, g2, b2, x2, xtb);

  // 10) transition: xt @ tr_win (N = 6144)
  dim3 gT(TOK_ / 64, (2 * INNER_) / 64);
  gemm_bf16<<<gT, 256, 0, stream>>>(xtb, t_win, nullptr, t1, TOK_, 2 * INNER_, DIM_);

  // 11) a * silu(g) -> bf16
  swiglu_kernel<<<1024, 256, 0, stream>>>(t1, hb);

  // 12) @ tr_wout (K = 3072)
  gemm_bf16<<<gC, 256, 0, stream>>>(hb, t_wout, nullptr, t2, TOK_, DIM_, INNER_);

  // 13) scale2 + residual -> out
  final_kernel<<<512, 256, 0, stream>>>(t2, s2, x2, mask, outp);
}
